// WordSpeechCtC_48567490183811
// MI455X (gfx1250) — compile-verified, matrix operand order checked by reading, not run on hardware
//
#include <hip/hip_runtime.h>

// Problem dims
#define B_  32
#define T_  2048
#define D_  1024
#define H_  1024
#define W_  48
#define NG  4096   // 4*H

typedef __attribute__((ext_vector_type(16))) __bf16 v16bf;
typedef __attribute__((ext_vector_type(8)))  float  v8f;
typedef int v4i __attribute__((ext_vector_type(4)));

// ---- CDNA5 async global->LDS (ASYNCcnt) path, feature-guarded --------------
// Probe-learned signature: param0 = AS1 (global) pointer to 4xi32 vector,
// param1 = LDS pointer, then offset + cpol immediates.

#if __has_builtin(__builtin_amdgcn_global_load_async_to_lds_b128)
#define HAS_ASYNC 1
#define GASYNC_B128(gsrc, ldst)                                            \
  __builtin_amdgcn_global_load_async_to_lds_b128(                          \
      (__attribute__((address_space(1))) v4i*)(void*)(const void*)(gsrc),  \
      (__attribute__((address_space(3))) v4i*)(void*)(ldst), 0, 0)
#if __has_builtin(__builtin_amdgcn_s_wait_asynccnt)
#define WAIT_ASYNC() __builtin_amdgcn_s_wait_asynccnt(0)
#else
#define WAIT_ASYNC() asm volatile("s_wait_asynccnt 0x0" ::: "memory")
#endif
#else
#define HAS_ASYNC 0
#endif

// ---- helpers ---------------------------------------------------------------

__device__ __forceinline__ unsigned short f2bf(float f) {
  unsigned u = __float_as_uint(f);
  u += 0x7FFFu + ((u >> 16) & 1u);           // round-to-nearest-even
  return (unsigned short)(u >> 16);
}

// Load one 16x32 bf16 WMMA fragment from an LDS tile.
// ISA 16-bit A/B layout: lane = row (M or N), 8 dwords of K-pairs:
// lanes 0-15 -> dwords {0..3, 8..11}, lanes 16-31 -> dwords {4..7, 12..15}.
__device__ __forceinline__ v16bf frag_ld(const unsigned* tile, int pitch,
                                         int koff, int lane) {
  const unsigned* row = tile + (lane & 15) * pitch + koff + ((lane >> 4) << 2);
  union { uint4 q[2]; v16bf v; } u;
  u.q[0] = *(const uint4*)(row);
  u.q[1] = *(const uint4*)(row + 8);
  return u.v;
}

// ---- init: convert weights to bf16, fuse bias, zero state/output -----------

__global__ void init_kernel(const float* __restrict__ wih_f,
                            const float* __restrict__ whh_f,
                            const float* __restrict__ b_ih,
                            const float* __restrict__ b_hh,
                            unsigned short* __restrict__ wih_bf,
                            unsigned short* __restrict__ whh_bf,
                            float* __restrict__ bias,
                            unsigned* __restrict__ hbuf_dw,
                            float* __restrict__ out,
                            unsigned* __restrict__ counter) {
  size_t i = (size_t)blockIdx.x * blockDim.x + threadIdx.x;
  if (i < (size_t)NG * D_) {
    wih_bf[i] = f2bf(wih_f[i]);
    whh_bf[i] = f2bf(whh_f[i]);
  }
  if (i < NG) bias[i] = b_ih[i] + b_hh[i];
  if (i < (size_t)2 * B_ * H_ / 2) hbuf_dw[i] = 0;      // zero both h buffers
  if (i < (size_t)B_ * W_ * H_ + B_) out[i] = 0.f;      // zero out + seq_len
  if (i == 0) *counter = 0;
}

// ---- convert frame_input (B*T, D) f32 -> bf16, vectorized ------------------

__global__ __launch_bounds__(256) void cvt_kernel(
    const float* __restrict__ src, unsigned short* __restrict__ dst) {
  size_t i = (size_t)blockIdx.x * blockDim.x + threadIdx.x;  // one float4 each
  const size_t n4 = (size_t)B_ * T_ * D_ / 4;
  if (i >= n4) return;
  float4 f = ((const float4*)src)[i];
  uint2 o;
  o.x = (unsigned)f2bf(f.x) | ((unsigned)f2bf(f.y) << 16);
  o.y = (unsigned)f2bf(f.z) | ((unsigned)f2bf(f.w) << 16);
  ((uint2*)dst)[i] = o;
}

// ---- segment table: -2 masked, -1 mid-segment, >=0 scatter slot ------------

__global__ void seg_kernel(const int* __restrict__ map,
                           int* __restrict__ slotTab,
                           float* __restrict__ seq_len) {
  int b = threadIdx.x;
  if (b >= B_) return;
  int cnt = 0;
  for (int t = 0; t < T_; ++t) {
    int w = map[b * T_ + t];
    int s;
    if (w == 0) {
      s = -2;
    } else {
      bool end = (t == T_ - 1) || (map[b * T_ + t + 1] != w);
      s = end ? cnt++ : -1;
    }
    slotTab[b * T_ + t] = s;
  }
  seq_len[b] = (float)cnt;
}

// ---- kernel 1: xproj = X (B*T x D) * W_ih^T + bias, bf16 WMMA --------------
// Block: 256 thr = 8 waves. Block tile 128(M) x 128(N), K chunks of 32,
// double-buffered LDS. Each wave owns a 64x32 sub-tile = 8 WMMA tiles
// (64 acc VGPRs + 48 fragment VGPRs: no spills).

#define APITCH 20   // dwords per LDS row: 16 data + 4 pad (rows 16B-aligned,
                    // stride 20 mod 64 -> conflict-free 16-row fragment reads)

__global__ __launch_bounds__(256) void xproj_kernel(
    const unsigned short* __restrict__ xbf,   // (B*T, D) bf16
    const unsigned short* __restrict__ wih,   // (NG, D)  bf16
    const float* __restrict__ bias,
    float* __restrict__ xproj) {
  __shared__ unsigned As[2][128 * APITCH];    // 2 x 10.0 KB
  __shared__ unsigned Bs[2][128 * APITCH];    // 2 x 10.0 KB

  const int tid  = threadIdx.x;
  const int lane = tid & 31;
  const int wid  = tid >> 5;
  const int wm = wid >> 2;                    // 64-row slab within block
  const int wn = wid & 3;                     // 32-col slab within block
  const int N0 = blockIdx.x * 128;
  const int M0 = blockIdx.y * 128;

  const uint4* aq = (const uint4*)xbf + (size_t)M0 * (D_ / 8);
  const uint4* bq = (const uint4*)wih + (size_t)N0 * (D_ / 8);

  v8f acc[8];
#pragma unroll
  for (int i = 0; i < 8; ++i) acc[i] = (v8f){};

#if !HAS_ASYNC
  uint4 ra[2], rb[2];                          // fallback staging registers
#endif

  auto issue_stage = [&](int kb, int buf) {    // global -> LDS buf (async) or
#if HAS_ASYNC                                  // global -> regs (fallback)
#pragma unroll
    for (int i = 0; i < 2; ++i) {
      int e = tid + i * 256;                   // A: 128 rows x 4 uint4
      GASYNC_B128(aq + (size_t)(e >> 2) * 128 + kb * 4 + (e & 3),
                  (uint4*)&As[buf][(e >> 2) * APITCH] + (e & 3));
    }
#pragma unroll
    for (int i = 0; i < 2; ++i) {
      int e = tid + i * 256;                   // B: 128 rows x 4 uint4
      GASYNC_B128(bq + (size_t)(e >> 2) * 128 + kb * 4 + (e & 3),
                  (uint4*)&Bs[buf][(e >> 2) * APITCH] + (e & 3));
    }
#else
    (void)buf;
#pragma unroll
    for (int i = 0; i < 2; ++i) {
      int e = tid + i * 256;
      ra[i] = aq[(size_t)(e >> 2) * 128 + kb * 4 + (e & 3)];
      rb[i] = bq[(size_t)(e >> 2) * 128 + kb * 4 + (e & 3)];
    }
#endif
  };
  auto commit_stage = [&](int buf) {           // make buf ready (pre-barrier)
#if HAS_ASYNC
    (void)buf;
    WAIT_ASYNC();
#else
#pragma unroll
    for (int i = 0; i < 2; ++i) {
      int e = tid + i * 256;
      ((uint4*)&As[buf][(e >> 2) * APITCH])[e & 3] = ra[i];
      ((uint4*)&Bs[buf][(e >> 2) * APITCH])[e & 3] = rb[i];
    }
#endif
  };

  issue_stage(0, 0);
  commit_stage(0);
  __syncthreads();

  for (int kb = 0; kb < 32; ++kb) {
    const int buf = kb & 1;
    if (kb + 1 < 32) issue_stage(kb + 1, buf ^ 1);

    v16bf af[4], bf[2];
#pragma unroll
    for (int i = 0; i < 4; ++i)
      af[i] = frag_ld(&As[buf][(wm * 64 + i * 16) * APITCH], APITCH, 0, lane);
#pragma unroll
    for (int j = 0; j < 2; ++j)
      bf[j] = frag_ld(&Bs[buf][(wn * 32 + j * 16) * APITCH], APITCH, 0, lane);
#pragma unroll
    for (int i = 0; i < 4; ++i)
#pragma unroll
      for (int j = 0; j < 2; ++j)
        acc[i * 2 + j] = __builtin_amdgcn_wmma_f32_16x16x32_bf16(
            false, af[i], false, bf[j], (short)0, acc[i * 2 + j], false,
            false);

    if (kb + 1 < 32) commit_stage(buf ^ 1);
    __syncthreads();
  }

  // C layout: lane<16 -> M=v, lane>=16 -> M=v+8; N = lane&15
#pragma unroll
  for (int j = 0; j < 2; ++j) {
    const int col = N0 + wn * 32 + j * 16 + (lane & 15);
    const float bv = bias[col];
#pragma unroll
    for (int i = 0; i < 4; ++i) {
      const int mb = M0 + wm * 64 + i * 16 + ((lane >> 4) << 3);
#pragma unroll
      for (int v = 0; v < 8; ++v)
        xproj[(size_t)(mb + v) * NG + col] = acc[i * 2 + j][v] + bv;
    }
  }
}

// ---- kernel 2: persistent masked LSTM recurrence ---------------------------
// 64 WGs; WG g owns hidden units [g*16, g*16+16). Its 64 W_hh rows
// (i/f/g/o x 16) stay resident in LDS for all 2048 steps. 8 waves =
// 2 batch-tiles x 4 gate-blocks, each accumulating one 16x16 tile over K=1024.

__global__ __launch_bounds__(256) void lstm_rec_kernel(
    const float* __restrict__ xproj, const unsigned short* __restrict__ whh,
    const int* __restrict__ slotTab, unsigned short* __restrict__ hbuf,
    float* __restrict__ out, unsigned* __restrict__ counter) {
  __shared__ unsigned Ws[64 * 516];   // 64 rows x (512 data + 4 pad) dw = 129KB
  __shared__ unsigned Hs[32 * 516];   // staged h(t), bf16, 64.5KB
  __shared__ float    Gs[4][32][16];  // gate exchange
  __shared__ float    Cs[32][16];     // cell state (resident all steps)

  const int tid  = threadIdx.x;
  const int lane = tid & 31;
  const int wid  = tid >> 5;
  const int mt = wid >> 2;            // batch tile 0..1
  const int gb = wid & 3;             // gate block i/f/g/o
  const int j0 = blockIdx.x * 16;

  // Stage this WG's W_hh slice once (row r -> W_hh[(r>>4)*H + j0 + (r&15)]).
  const uint4* whh_q = (const uint4*)whh;
#if HAS_ASYNC
#pragma unroll
  for (int i = 0; i < 32; ++i) {
    int e = tid + i * 256;
    int r = e >> 7;
    int grow = (r >> 4) * H_ + j0 + (r & 15);
    GASYNC_B128(whh_q + (size_t)grow * 128 + (e & 127),
                (uint4*)&Ws[r * 516] + (e & 127));
  }
  WAIT_ASYNC();
#else
#pragma unroll
  for (int grp = 0; grp < 4; ++grp) {
    uint4 tmp[8];
#pragma unroll
    for (int i = 0; i < 8; ++i) {
      int e = tid + (grp * 8 + i) * 256;
      int grow = ((e >> 7) >> 4) * H_ + j0 + ((e >> 7) & 15);
      tmp[i] = whh_q[(size_t)grow * 128 + (e & 127)];
    }
#pragma unroll
    for (int i = 0; i < 8; ++i) {
      int e = tid + (grp * 8 + i) * 256;
      ((uint4*)&Ws[(e >> 7) * 516])[e & 127] = tmp[i];
    }
  }
#endif
  for (int e = tid; e < 512; e += 256) ((float*)Cs)[e] = 0.f;
  __syncthreads();

  for (int t = 0; t < T_; ++t) {
    // stage h(t) (bf16, double-buffered in global)
    const uint4* hq = (const uint4*)(hbuf + (size_t)(t & 1) * B_ * H_);
#if HAS_ASYNC
#pragma unroll
    for (int i = 0; i < 16; ++i) {
      int e = tid + i * 256;
      GASYNC_B128(hq + (size_t)(e >> 7) * 128 + (e & 127),
                  (uint4*)&Hs[(e >> 7) * 516] + (e & 127));
    }
    WAIT_ASYNC();
#else
#pragma unroll
    for (int half = 0; half < 2; ++half) {
      uint4 tmp[8];
#pragma unroll
      for (int i = 0; i < 8; ++i) {
        int e = tid + (half * 8 + i) * 256;
        tmp[i] = hq[(size_t)(e >> 7) * 128 + (e & 127)];
      }
#pragma unroll
      for (int i = 0; i < 8; ++i) {
        int e = tid + (half * 8 + i) * 256;
        ((uint4*)&Hs[(e >> 7) * 516])[e & 127] = tmp[i];
      }
    }
#endif
    __syncthreads();

    // init accumulator from precomputed x-projection (bias folded in)
    const int col   = gb * H_ + j0 + (lane & 15);
    const int bbase = mt * 16 + ((lane >> 4) << 3);
    v8f acc;
#pragma unroll
    for (int v = 0; v < 8; ++v)
      acc[v] = xproj[((size_t)(bbase + v) * T_ + t) * (size_t)NG + col];

    // gates += h(t) * W_hh^T  (K = 1024, 32 WMMAs)
#pragma unroll 8
    for (int kb = 0; kb < 32; ++kb) {
      v16bf a = frag_ld(&Hs[(mt * 16) * 516], 516, kb * 16, lane);
      v16bf b = frag_ld(&Ws[(gb * 16) * 516], 516, kb * 16, lane);
      acc = __builtin_amdgcn_wmma_f32_16x16x32_bf16(
          false, a, false, b, (short)0, acc, false, false);
    }

    // exchange gate tiles through LDS so each thread sees i/f/g/o per unit
    {
      int jj = lane & 15;
      int br = mt * 16 + ((lane >> 4) << 3);
#pragma unroll
      for (int v = 0; v < 8; ++v) Gs[gb][br + v][jj] = acc[v];
    }
    __syncthreads();

    // LSTM cell update for 512 (batch, unit) pairs; write h(t+1)
    unsigned short* hnext = hbuf + (size_t)((t + 1) & 1) * B_ * H_;
#pragma unroll
    for (int e = tid; e < 512; e += 256) {
      int b = e >> 4, jj = e & 15;
      int st = slotTab[b * T_ + t];
      unsigned hw = Hs[b * 516 + ((j0 + jj) >> 1)];
      unsigned short hstore = (unsigned short)((jj & 1) ? (hw >> 16) : hw);
      if (st != -2) {  // frame not masked: advance state
        float iv = Gs[0][b][jj], fv = Gs[1][b][jj];
        float gv = Gs[2][b][jj], ov = Gs[3][b][jj];
        float c_old = Cs[b][jj];
        float ig = 1.f / (1.f + __expf(-iv));
        float fg = 1.f / (1.f + __expf(-fv));
        float og = 1.f / (1.f + __expf(-ov));
        float c_new = fg * c_old + ig * tanhf(gv);
        float h_new = og * tanhf(c_new);
        Cs[b][jj] = c_new;
        hstore = f2bf(h_new);
        if (st >= 0)  // last frame of a word: scatter fp32 embedding
          out[((size_t)b * W_ + st) * H_ + j0 + jj] = h_new;
      }
      hnext[(size_t)b * H_ + j0 + jj] = hstore;
    }

    // grid-wide barrier between timesteps (release h(t+1), acquire peers')
    __threadfence();
    __syncthreads();
    if (tid == 0) {
      __hip_atomic_fetch_add(counter, 1u, __ATOMIC_ACQ_REL,
                             __HIP_MEMORY_SCOPE_AGENT);
      unsigned target = (unsigned)gridDim.x * (unsigned)(t + 1);
      while (__hip_atomic_load(counter, __ATOMIC_ACQUIRE,
                               __HIP_MEMORY_SCOPE_AGENT) < target)
        __builtin_amdgcn_s_sleep(2);
    }
    __syncthreads();
  }
}

// ---- launch ----------------------------------------------------------------

extern "C" void kernel_launch(void* const* d_in, const int* in_sizes, int n_in,
                              void* d_out, int out_size, void* d_ws,
                              size_t ws_size, hipStream_t stream) {
  const float* frame = (const float*)d_in[0];
  const float* W_ih  = (const float*)d_in[1];
  const float* W_hh  = (const float*)d_in[2];
  const float* b_ih  = (const float*)d_in[3];
  const float* b_hh  = (const float*)d_in[4];
  const int*   map   = (const int*)d_in[5];
  float* out = (float*)d_out;

  char* ws = (char*)d_ws;
  auto take = [&](size_t bytes) {
    char* p = ws;
    ws += (bytes + 4095) & ~(size_t)4095;
    return p;
  };
  unsigned short* wih_bf  = (unsigned short*)take((size_t)NG * D_ * 2);
  unsigned short* whh_bf  = (unsigned short*)take((size_t)NG * H_ * 2);
  float*          bias    = (float*)take((size_t)NG * 4);
  unsigned short* hbuf    = (unsigned short*)take((size_t)2 * B_ * H_ * 2);
  unsigned*       counter = (unsigned*)take(4096);
  int*            slotTab = (int*)take((size_t)B_ * T_ * 4);
  unsigned short* xbf     = (unsigned short*)take((size_t)B_ * T_ * D_ * 2);
  float*          xproj   = (float*)take((size_t)B_ * T_ * NG * 4);

  init_kernel<<<16384, 256, 0, stream>>>(W_ih, W_hh, b_ih, b_hh, wih_bf,
                                         whh_bf, bias, (unsigned*)hbuf, out,
                                         counter);
  cvt_kernel<<<(B_ * T_ * D_ / 4 + 255) / 256, 256, 0, stream>>>(frame, xbf);
  seg_kernel<<<1, 32, 0, stream>>>(map, slotTab, out + (size_t)B_ * W_ * H_);
  dim3 g1(NG / 128, (B_ * T_) / 128);
  xproj_kernel<<<g1, 256, 0, stream>>>(xbf, wih_bf, bias, xproj);
  lstm_rec_kernel<<<64, 256, 0, stream>>>(xproj, whh_bf, slotTab, hbuf, out,
                                          counter);
}
